// Linear_1357209666318
// MI455X (gfx1250) — 8-bit matrix operand order repaired and checked by reading, compiled, not run on hardware
//
#include <hip/hip_runtime.h>
#include <hip/hip_bf16.h>
#include <stdint.h>

typedef __attribute__((ext_vector_type(16))) int          v16i;
typedef __attribute__((ext_vector_type(8)))  float        v8f;
typedef __attribute__((ext_vector_type(4)))  float        f32x4;
typedef __attribute__((ext_vector_type(2)))  unsigned int u32x2;
typedef __attribute__((ext_vector_type(4)))  unsigned int u32x4;

#define M_DIM 4096
#define N_DIM 4096
#define K_DIM 4096
#define BLK   128
#define KB    (K_DIM / BLK)
#define FP8_MAX 448.0f

#define LDS_STRIDE 144
#define TILE_BYTES (128 * LDS_STRIDE)
#define LDS_TOTAL  (2 * 2 * TILE_BYTES)

__device__ inline unsigned int f32_to_e4m3_sw(float f) {
    unsigned int u = __float_as_uint(f);
    unsigned int sign = (u >> 24) & 0x80u;
    float a = fabsf(f);
    if (a > 448.0f) a = 448.0f;
    if (a < 0.0078125f) return sign;
    unsigned int x = __float_as_uint(a);
    unsigned int lsb = (x >> 20) & 1u;
    x += 0x0007FFFFu + lsb;
    int e = (int)((x >> 23) & 0xFFu) - 127;
    unsigned int m = (x >> 20) & 7u;
    if (e < -6) return sign;
    return sign | ((unsigned int)(e + 7) << 3) | m;
}

__device__ inline unsigned int pack4_fp8(float a, float b, float c, float d) {
#if __has_builtin(__builtin_amdgcn_cvt_pk_fp8_f32)
    int r = 0;
    r = __builtin_amdgcn_cvt_pk_fp8_f32(a, b, r, false);
    r = __builtin_amdgcn_cvt_pk_fp8_f32(c, d, r, true);
    return (unsigned int)r;
#else
    return f32_to_e4m3_sw(a) | (f32_to_e4m3_sw(b) << 8) |
           (f32_to_e4m3_sw(c) << 16) | (f32_to_e4m3_sw(d) << 24);
#endif
}

__device__ inline void async_load_b128(const void* gsrc, unsigned lds_byte_off) {
    asm volatile("global_load_async_to_lds_b128 %0, %1, off"
                 :: "v"(lds_byte_off), "v"(gsrc)
                 : "memory");
}

template <int N>
__device__ inline void wait_asynccnt() {
#if __has_builtin(__builtin_amdgcn_s_wait_asynccnt)
    __builtin_amdgcn_s_wait_asynccnt(N);
#else
    if (N == 0) asm volatile("s_wait_asynccnt 0x0" ::: "memory");
    else        asm volatile("s_wait_asynccnt 0x8" ::: "memory");
#endif
}

__device__ inline void async_copy_tile(const unsigned char* __restrict__ g,
                                       unsigned lds_off, int tid) {
#pragma unroll
    for (int i = 0; i < 4; ++i) {
        const int c   = tid + i * 256;
        const int row = c >> 3;
        const int col = (c & 7) << 4;
        async_load_b128(g + (size_t)row * K_DIM + col,
                        lds_off + (unsigned)(row * LDS_STRIDE + col));
    }
}

__global__ __launch_bounds__(256) void act_quant_kernel(
        const float* __restrict__ x, unsigned char* __restrict__ xq,
        float* __restrict__ xs_t  ) {
    const int lane = threadIdx.x & 31;
    const int wave = threadIdx.x >> 5;
    const int task = blockIdx.x * 8 + wave;
    const int row  = task / KB;
    const int kb   = task % KB;
    const size_t base = (size_t)row * K_DIM + (size_t)kb * BLK + (size_t)lane * 4;
    f32x4 v = *(const f32x4*)(x + base);
    float amax = fmaxf(fmaxf(fabsf(v.x), fabsf(v.y)), fmaxf(fabsf(v.z), fabsf(v.w)));
#pragma unroll
    for (int off = 16; off >= 1; off >>= 1)
        amax = fmaxf(amax, __shfl_xor(amax, off, 32));
    float s   = amax * (1.0f / FP8_MAX);
    float inv = (amax > 0.0f) ? (FP8_MAX / amax) : 0.0f;
    *(unsigned int*)(xq + base) = pack4_fp8(v.x * inv, v.y * inv, v.z * inv, v.w * inv);
    if (lane == 0) xs_t[(size_t)kb * M_DIM + row] = s;
}

__global__ __launch_bounds__(256) void wq_cvt_kernel(
        const float* __restrict__ w, unsigned char* __restrict__ wq) {
    const size_t i = (size_t)blockIdx.x * 256 + threadIdx.x;
    f32x4 v = *(const f32x4*)(w + i * 4);
    ((unsigned int*)wq)[i] = pack4_fp8(v.x, v.y, v.z, v.w);
}

__global__ __launch_bounds__(256) void gemm_fp8_kernel(
        const unsigned char* __restrict__ xq,
        const unsigned char* __restrict__ wq,
        const float* __restrict__ xs_t,
        const float* __restrict__ ws,
        float* __restrict__ out) {
    extern __shared__ unsigned char smem[];

    const int tid  = threadIdx.x;
    const int lane = tid & 31;
    const int wave = tid >> 5;
    const int wm   = wave & 3;
    const int wn   = wave >> 2;
    const int hi   = lane >> 4;
    const int ln   = lane & 15;

    const int m_wg = blockIdx.y * 128;
    const int n_wg = blockIdx.x * 128;

    const unsigned char* Ag = xq + (size_t)m_wg * K_DIM;
    const unsigned char* Bg = wq + (size_t)n_wg * K_DIM;

    v8f acc[2][4];
#pragma unroll
    for (int mt = 0; mt < 2; ++mt)
#pragma unroll
        for (int nt = 0; nt < 4; ++nt) {
            v8f z = {0.f, 0.f, 0.f, 0.f, 0.f, 0.f, 0.f, 0.f};
            acc[mt][nt] = z;
        }

    const unsigned aRow0 = (unsigned)((wm * 32 + ln) * LDS_STRIDE + hi * 16);
    const unsigned bRow0 = (unsigned)((wn * 64 + ln) * LDS_STRIDE + hi * 16);

    const float* asp = xs_t + m_wg + wm * 32 + hi * 8;
    const float* wsp = ws + (size_t)blockIdx.x * KB;

    async_copy_tile(Ag, 0u, tid);
    async_copy_tile(Bg, (unsigned)TILE_BYTES, tid);

    for (int kb = 0; kb < KB; ++kb) {
        const int buf = kb & 1;
        const unsigned bufA = (unsigned)(buf * 2 * TILE_BYTES);
        const unsigned bufB = bufA + (unsigned)TILE_BYTES;

        if (kb + 1 < KB) {
            const unsigned nA = (unsigned)((buf ^ 1) * 2 * TILE_BYTES);
            async_copy_tile(Ag + (kb + 1) * BLK, nA, tid);
            async_copy_tile(Bg + (kb + 1) * BLK, nA + (unsigned)TILE_BYTES, tid);
            wait_asynccnt<8>();
        } else {
            wait_asynccnt<0>();
        }
        __syncthreads();

        v16i a[2];
#pragma unroll
        for (int mt = 0; mt < 2; ++mt) {
            const unsigned char* p = smem + bufA + aRow0 + mt * 16 * LDS_STRIDE;
#pragma unroll
            for (int g = 0; g < 4; ++g) {
                u32x4 t = *(const u32x4*)(p + g * 32);
                a[mt][4 * g]     = (int)t.x;
                a[mt][4 * g + 1] = (int)t.y;
                a[mt][4 * g + 2] = (int)t.z;
                a[mt][4 * g + 3] = (int)t.w;
            }
        }
        v16i b[4];
#pragma unroll
        for (int nt = 0; nt < 4; ++nt) {
            const unsigned char* p = smem + bufB + bRow0 + nt * 16 * LDS_STRIDE;
#pragma unroll
            for (int g = 0; g < 4; ++g) {
                u32x4 t = *(const u32x4*)(p + g * 32);
                b[nt][4 * g]     = (int)t.x;
                b[nt][4 * g + 1] = (int)t.y;
                b[nt][4 * g + 2] = (int)t.z;
                b[nt][4 * g + 3] = (int)t.w;
            }
        }

        const float bs = wsp[kb];
        f32x4 s0a = *(const f32x4*)(asp);
        f32x4 s0b = *(const f32x4*)(asp + 4);
        f32x4 s1a = *(const f32x4*)(asp + 16);
        f32x4 s1b = *(const f32x4*)(asp + 20);
        float sc[2][8];
        sc[0][0] = s0a.x * bs; sc[0][1] = s0a.y * bs; sc[0][2] = s0a.z * bs; sc[0][3] = s0a.w * bs;
        sc[0][4] = s0b.x * bs; sc[0][5] = s0b.y * bs; sc[0][6] = s0b.z * bs; sc[0][7] = s0b.w * bs;
        sc[1][0] = s1a.x * bs; sc[1][1] = s1a.y * bs; sc[1][2] = s1a.z * bs; sc[1][3] = s1a.w * bs;
        sc[1][4] = s1b.x * bs; sc[1][5] = s1b.y * bs; sc[1][6] = s1b.z * bs; sc[1][7] = s1b.w * bs;

#pragma unroll
        for (int mt = 0; mt < 2; ++mt)
#pragma unroll
            for (int nt = 0; nt < 4; ++nt) {
                v8f c = {0.f, 0.f, 0.f, 0.f, 0.f, 0.f, 0.f, 0.f};
                c = __builtin_amdgcn_wmma_f32_16x16x128_fp8_fp8(
                        a[mt], b[nt], (short)0, c, false, false);
#pragma unroll
                for (int r = 0; r < 8; ++r)
                    acc[mt][nt][r] = fmaf(c[r], sc[mt][r], acc[mt][nt][r]);
            }
        asp += M_DIM;

        __syncthreads();
    }

#pragma unroll
    for (int mt = 0; mt < 2; ++mt)
#pragma unroll
        for (int r = 0; r < 8; ++r) {
            const size_t row = (size_t)(m_wg + wm * 32 + mt * 16 + hi * 8 + r);
#pragma unroll
            for (int nt = 0; nt < 4; ++nt) {
                const int col = n_wg + wn * 64 + nt * 16 + ln;
                out[row * N_DIM + col] = acc[mt][nt][r];
            }
        }
}

extern "C" void kernel_launch(void* const* d_in, const int* in_sizes, int n_in,
                              void* d_out, int out_size, void* d_ws, size_t ws_size,
                              hipStream_t stream) {
    const float* x   = (const float*)d_in[0];
    const float* wqf = (const float*)d_in[1];
    const float* ws  = (const float*)d_in[2];
    float* out = (float*)d_out;

    unsigned char* xq8 = (unsigned char*)d_ws;
    unsigned char* wq8 = xq8 + (size_t)M_DIM * K_DIM;
    float* xs_t = (float*)(wq8 + (size_t)N_DIM * K_DIM);

    act_quant_kernel<<<(M_DIM * KB) / 8, 256, 0, stream>>>(x, xq8, xs_t);
    wq_cvt_kernel<<<((size_t)N_DIM * K_DIM / 4) / 256, 256, 0, stream>>>(wqf, wq8);

    dim3 grid(N_DIM / 128, M_DIM / 128);
    gemm_fp8_kernel<<<grid, 256, LDS_TOTAL, stream>>>(xq8, wq8, xs_t, ws, out);
}
